// PALM_70763881169526
// MI455X (gfx1250) — matrix operand order repaired and checked by reading, not compiled, not run on hardware
//
#include <hip/hip_runtime.h>

typedef __attribute__((ext_vector_type(16))) __bf16 v16bf;
typedef __attribute__((ext_vector_type(8)))  float  v8f;

constexpr int   M_ROWS   = 32768;   // BATCH * NVIEWS
constexpr int   NPROT    = 1024;    // N_PROTOS
constexpr int   FDIM     = 128;     // FEAT_DIM
constexpr int   BATCH_N  = 16384;
constexpr float INV_EPS  = 20.0f;   // 1/0.05
constexpr float INV_TEMP = 10.0f;   // 1/0.1
constexpr float INV_PTMP = 10.0f;   // 1/0.1
constexpr int   KTOP     = 5;

// ---------------- protos: L2-normalize + bf16 convert ----------------
__global__ void k_norm_protos(const float* __restrict__ protos, __bf16* __restrict__ pb) {
  __shared__ float red[128];
  int r = blockIdx.x, t = threadIdx.x;
  float x = protos[r * FDIM + t];
  red[t] = x * x; __syncthreads();
  for (int o = 64; o > 0; o >>= 1) { if (t < o) red[t] += red[t + o]; __syncthreads(); }
  float inv = 1.0f / fmaxf(sqrtf(red[0]), 1e-12f);
  pb[r * FDIM + t] = (__bf16)(x * inv);
}

__global__ void k_f32_to_bf16(const float* __restrict__ in, __bf16* __restrict__ out, int n) {
  int i = blockIdx.x * blockDim.x + threadIdx.x;
  if (i < n) out[i] = (__bf16)in[i];
}

// ---------------- WMMA GEMM: C[M,N] = A[M,128] * B[N,128]^T * scale ----------------
// 8 waves/block; wave = 16 rows x 128 cols; K-loop 4 chunks of 32 (bf16 WMMA, f32 acc)
__global__ void __launch_bounds__(256) k_gemm_bt(const __bf16* __restrict__ A,
                                                 const __bf16* __restrict__ B,
                                                 float* __restrict__ C, int ldc, float scale) {
  int wave = threadIdx.x >> 5, lane = threadIdx.x & 31;
  int half = lane >> 4, l = lane & 15;
  int mr = blockIdx.x * 128 + wave * 16;
  int nc = blockIdx.y * 128;
  v8f acc[8] = {};
  const __bf16* arow = A + (size_t)(mr + l) * FDIM;
#pragma unroll
  for (int kc = 0; kc < FDIM; kc += 32) {
    union { v16bf v; int4 i[2]; } af;
    af.i[0] = *(const int4*)(arow + kc + half * 8);        // K = kc+0..7  (+8 for upper half)
    af.i[1] = *(const int4*)(arow + kc + 16 + half * 8);   // K = kc+16..23 (+8)
    __builtin_prefetch(arow + kc + 32, 0, 0);              // global_prefetch_b8
#pragma unroll
    for (int j = 0; j < 8; ++j) {
      const __bf16* brow = B + (size_t)(nc + j * 16 + l) * FDIM;
      union { v16bf v; int4 i[2]; } bfv;
      bfv.i[0] = *(const int4*)(brow + kc + half * 8);      // K = kc+0..7  / kc+8..15
      bfv.i[1] = *(const int4*)(brow + kc + 16 + half * 8); // K = kc+16..23 / kc+24..31
      acc[j] = __builtin_amdgcn_wmma_f32_16x16x32_bf16(
          false, af.v, false, bfv.v, (short)0, acc[j], false, false);
    }
  }
#pragma unroll
  for (int j = 0; j < 8; ++j) {
    int col = nc + j * 16 + l;
#pragma unroll
    for (int r = 0; r < 8; ++r) {
      int row = mr + half * 8 + r;
      C[(size_t)row * ldc + col] = acc[j][r] * scale;
    }
  }
}

// ---------------- rowmax + E = exp((S-rowmax)/eps), per-row sum ----------------
__global__ void k_rowmax_exp(const float* __restrict__ S, float* __restrict__ E,
                             float* __restrict__ rowsum) {
  __shared__ float red[256];
  int row = blockIdx.x, t = threadIdx.x;
  const float* sr = S + (size_t)row * NPROT;
  float s[4]; float m = -1e30f;
#pragma unroll
  for (int q = 0; q < 4; ++q) { s[q] = sr[t + q * 256]; m = fmaxf(m, s[q]); }
  red[t] = m; __syncthreads();
  for (int o = 128; o > 0; o >>= 1) { if (t < o) red[t] = fmaxf(red[t], red[t + o]); __syncthreads(); }
  float rm = red[0]; __syncthreads();
  float* er = E + (size_t)row * NPROT;
  float es = 0.f;
#pragma unroll
  for (int q = 0; q < 4; ++q) { float e = __expf((s[q] - rm) * INV_EPS); er[t + q * 256] = e; es += e; }
  red[t] = es; __syncthreads();
  for (int o = 128; o > 0; o >>= 1) { if (t < o) red[t] += red[t + o]; __syncthreads(); }
  if (t == 0) rowsum[row] = red[0];
}

__global__ void k_reduce_total(const float* __restrict__ rowsum, float* __restrict__ scal) {
  __shared__ float red[256];
  int t = threadIdx.x; float s = 0.f;
  for (int i = t; i < M_ROWS; i += 256) s += rowsum[i];
  red[t] = s; __syncthreads();
  for (int o = 128; o > 0; o >>= 1) { if (t < o) red[t] += red[t + o]; __syncthreads(); }
  if (t == 0) scal[0] = red[0];
}

__global__ void k_init_uv(float* __restrict__ u, float* __restrict__ v,
                          const float* __restrict__ scal) {
  int i = blockIdx.x * 256 + threadIdx.x;
  float inv_total = 1.0f / fmaxf(scal[0], 1e-30f);
  if (i < M_ROWS) v[i] = inv_total;
  if (i < NPROT)  u[i] = 1.0f;
}

// ---- Sinkhorn in scaling form: u-pass = weighted column sums of E ----
__global__ void k_colsum_part(const float* __restrict__ E, const float* __restrict__ v,
                              float* __restrict__ part) {
  int blk = blockIdx.x, t = threadIdx.x;
  float loc[4] = {0.f, 0.f, 0.f, 0.f};
  for (int r = 0; r < 128; ++r) {
    int row = blk * 128 + r;
    float vb = v[row];
    const float* er = E + (size_t)row * NPROT;
#pragma unroll
    for (int q = 0; q < 4; ++q) loc[q] += er[t + q * 256] * vb;
  }
#pragma unroll
  for (int q = 0; q < 4; ++q) part[(size_t)blk * NPROT + t + q * 256] = loc[q];
}

__global__ void k_finish_u(const float* __restrict__ part, float* __restrict__ u) {
  int c = blockIdx.x * 256 + threadIdx.x;
  float s = 0.f;
  for (int b = 0; b < 256; ++b) s += part[(size_t)b * NPROT + c];
  float uo = u[c];
  float rs = uo * s;                       // row sum in Qt domain
  u[c] = uo / (fmaxf(rs, 1e-12f) * (float)NPROT);
}

// ---- v-pass = weighted row sums of E ----
__global__ void k_update_v(const float* __restrict__ E, const float* __restrict__ u,
                           float* __restrict__ v) {
  __shared__ float red[256];
  int row = blockIdx.x, t = threadIdx.x;
  const float* er = E + (size_t)row * NPROT;
  float s = 0.f;
#pragma unroll
  for (int q = 0; q < 4; ++q) { int c = t + q * 256; s += er[c] * u[c]; }
  red[t] = s; __syncthreads();
  for (int o = 128; o > 0; o >>= 1) { if (t < o) red[t] += red[t + o]; __syncthreads(); }
  if (t == 0) {
    float vo = v[row];
    float cs = vo * red[0];
    v[row] = vo / (fmaxf(cs, 1e-12f) * (float)M_ROWS);
  }
}

// ---- per-row mle: logsumexp + top-5 of mask*u*E (row-scale invariant) ----
__global__ void k_mle_row(const float* __restrict__ S, const float* __restrict__ E,
                          const float* __restrict__ u, const int* __restrict__ targets,
                          float* __restrict__ out_row) {
  __shared__ float red[256];
  __shared__ float sv[256]; __shared__ int si[256];
  __shared__ float s_pick_logit;
  int row = blockIdx.x, t = threadIdx.x;
  int label = targets[row & (BATCH_N - 1)];
  const float* sr = S + (size_t)row * NPROT;
  const float* er = E + (size_t)row * NPROT;
  float lg[4], w[4];
  float m = -1e30f;
#pragma unroll
  for (int q = 0; q < 4; ++q) {
    int c = t + q * 256;
    float lv = sr[c] * INV_TEMP;
    lg[q] = lv; m = fmaxf(m, lv);
    w[q] = ((c & 1) == label) ? er[c] * u[c] : 0.0f;   // proto label = c % NUM_CLASSES
  }
  red[t] = m; __syncthreads();
  for (int o = 128; o > 0; o >>= 1) { if (t < o) red[t] = fmaxf(red[t], red[t + o]); __syncthreads(); }
  float rm = red[0]; __syncthreads();
  float es = 0.f;
#pragma unroll
  for (int q = 0; q < 4; ++q) es += __expf(lg[q] - rm);
  red[t] = es; __syncthreads();
  for (int o = 128; o > 0; o >>= 1) { if (t < o) red[t] += red[t + o]; __syncthreads(); }
  float neg = rm + __logf(red[0]);
  __syncthreads();
  float sw = 0.f, swl = 0.f;
  for (int it = 0; it < KTOP; ++it) {
    float bv = -1.0f; int bi = 0x7fffffff;
#pragma unroll
    for (int q = 0; q < 4; ++q) { if (w[q] > bv) { bv = w[q]; bi = t + q * 256; } }
    sv[t] = bv; si[t] = bi; __syncthreads();
    for (int o = 128; o > 0; o >>= 1) {
      if (t < o) {
        float ov = sv[t + o]; int oi = si[t + o];
        if (ov > sv[t] || (ov == sv[t] && oi < si[t])) { sv[t] = ov; si[t] = oi; }
      }
      __syncthreads();
    }
    int pick = si[0];
#pragma unroll
    for (int q = 0; q < 4; ++q) {
      if (t + q * 256 == pick) { s_pick_logit = lg[q]; w[q] = -1.0f; }
    }
    __syncthreads();
    if (t == 0) { sw += sv[0]; swl += sv[0] * s_pick_logit; }
    __syncthreads();
  }
  if (t == 0) out_row[row] = swl / fmaxf(sw, 1e-12f) - neg;
}

// ---- proto-contra per-row stats over adc[1024,1024] ----
__global__ void k_pc_row(const float* __restrict__ adc, float* __restrict__ out_row) {
  __shared__ float red[256];
  int row = blockIdx.x, t = threadIdx.x;
  int li = row & 1;
  const float* ar = adc + (size_t)row * NPROT;
  float a[4]; float m = -1e30f;
#pragma unroll
  for (int q = 0; q < 4; ++q) { a[q] = ar[t + q * 256]; m = fmaxf(m, a[q]); }
  red[t] = m; __syncthreads();
  for (int o = 128; o > 0; o >>= 1) { if (t < o) red[t] = fmaxf(red[t], red[t + o]); __syncthreads(); }
  float rm = red[0]; __syncthreads();
  float possum = 0.f, msum = 0.f, negsum = 0.f;
#pragma unroll
  for (int q = 0; q < 4; ++q) {
    int c = t + q * 256;
    float lv = a[q] - rm;
    bool diag = (c == row);
    float mk = (!diag && ((c & 1) == li)) ? 1.0f : 0.0f;
    possum += mk * lv;
    msum += mk;
    negsum += diag ? 0.0f : __expf(lv);
  }
  red[t] = possum; __syncthreads();
  for (int o = 128; o > 0; o >>= 1) { if (t < o) red[t] += red[t + o]; __syncthreads(); }
  possum = red[0]; __syncthreads();
  red[t] = msum; __syncthreads();
  for (int o = 128; o > 0; o >>= 1) { if (t < o) red[t] += red[t + o]; __syncthreads(); }
  msum = red[0]; __syncthreads();
  red[t] = negsum; __syncthreads();
  for (int o = 128; o > 0; o >>= 1) { if (t < o) red[t] += red[t + o]; __syncthreads(); }
  if (t == 0) {
    float pos = possum / fmaxf(msum, 1e-12f);
    out_row[row] = pos - __logf(red[0]);
  }
}

__global__ void k_final(const float* __restrict__ mle_row, const float* __restrict__ pc_row,
                        float* __restrict__ out) {
  __shared__ float red[256];
  int t = threadIdx.x;
  float s1 = 0.f;
  for (int i = t; i < M_ROWS; i += 256) s1 += mle_row[i];
  red[t] = s1; __syncthreads();
  for (int o = 128; o > 0; o >>= 1) { if (t < o) red[t] += red[t + o]; __syncthreads(); }
  s1 = red[0]; __syncthreads();
  float s2 = 0.f;
  for (int i = t; i < NPROT; i += 256) s2 += pc_row[i];
  red[t] = s2; __syncthreads();
  for (int o = 128; o > 0; o >>= 1) { if (t < o) red[t] += red[t + o]; __syncthreads(); }
  if (t == 0) out[0] = -(s1 / (float)M_ROWS) - (red[0] / (float)NPROT);  // LAMBDA_PCON = 1
}

extern "C" void kernel_launch(void* const* d_in, const int* in_sizes, int n_in,
                              void* d_out, int out_size, void* d_ws, size_t ws_size,
                              hipStream_t stream) {
  (void)in_sizes; (void)n_in; (void)out_size; (void)ws_size;
  const float* features = (const float*)d_in[0];
  const float* protos   = (const float*)d_in[1];
  const int*   targets  = (const int*)d_in[2];

  char* ws = (char*)d_ws;
  size_t off = 0;
  auto alloc = [&](size_t bytes) -> void* {
    size_t o = (off + 255) & ~(size_t)255;
    off = o + bytes;
    return (void*)(ws + o);
  };
  __bf16* featb  = (__bf16*)alloc((size_t)M_ROWS * FDIM * sizeof(__bf16));
  __bf16* protob = (__bf16*)alloc((size_t)NPROT * FDIM * sizeof(__bf16));
  float* S      = (float*)alloc((size_t)M_ROWS * NPROT * sizeof(float));
  float* E      = (float*)alloc((size_t)M_ROWS * NPROT * sizeof(float));
  float* rowsum = (float*)alloc((size_t)M_ROWS * sizeof(float));
  float* u      = (float*)alloc((size_t)NPROT * sizeof(float));
  float* v      = (float*)alloc((size_t)M_ROWS * sizeof(float));
  float* part   = (float*)alloc((size_t)256 * NPROT * sizeof(float));
  float* scal   = (float*)alloc(256);
  float* mle    = (float*)alloc((size_t)M_ROWS * sizeof(float));
  float* adc    = (float*)alloc((size_t)NPROT * NPROT * sizeof(float));
  float* pcr    = (float*)alloc((size_t)NPROT * sizeof(float));

  k_norm_protos<<<NPROT, 128, 0, stream>>>(protos, protob);
  k_f32_to_bf16<<<(M_ROWS * FDIM) / 256, 256, 0, stream>>>(features, featb, M_ROWS * FDIM);

  // S = feat @ protos_n^T   (raw dot; /TEMP applied downstream)
  k_gemm_bt<<<dim3(M_ROWS / 128, NPROT / 128), 256, 0, stream>>>(featb, protob, S, NPROT, 1.0f);

  k_rowmax_exp<<<M_ROWS, 256, 0, stream>>>(S, E, rowsum);
  k_reduce_total<<<1, 256, 0, stream>>>(rowsum, scal);
  k_init_uv<<<(M_ROWS + 255) / 256, 256, 0, stream>>>(u, v, scal);

  // 3 Sinkhorn iterations in scaling form; final v-update is provably unused by the loss
  for (int it = 0; it < 3; ++it) {
    k_colsum_part<<<256, 256, 0, stream>>>(E, v, part);
    k_finish_u<<<NPROT / 256, 256, 0, stream>>>(part, u);
    if (it < 2) k_update_v<<<M_ROWS, 256, 0, stream>>>(E, u, v);
  }

  k_mle_row<<<M_ROWS, 256, 0, stream>>>(S, E, u, targets, mle);

  // adc = (pn @ pn^T) / PROTO_TEMP
  k_gemm_bt<<<dim3(NPROT / 128, NPROT / 128), 256, 0, stream>>>(protob, protob, adc, NPROT, INV_PTMP);
  k_pc_row<<<NPROT, 256, 0, stream>>>(adc, pcr);

  k_final<<<1, 256, 0, stream>>>(mle, pcr, (float*)d_out);
}
